// GraphFusionLayer_36636071035402
// MI455X (gfx1250) — compile-verified, matrix operand order checked by reading, not run on hardware
//
#include <hip/hip_runtime.h>

#define DEV __device__ __forceinline__

typedef __attribute__((ext_vector_type(8)))  __bf16 bf16x8;
typedef __attribute__((ext_vector_type(16))) __bf16 bf16x16;
typedef __attribute__((ext_vector_type(8)))  float  f32x8;

DEV unsigned short f2bf(float f) {
    unsigned int u = __float_as_uint(f);
    u += 0x7FFFu + ((u >> 16) & 1u);          // round-to-nearest-even
    return (unsigned short)(u >> 16);
}
DEV float bf2f(unsigned short h) { return __uint_as_float(((unsigned int)h) << 16); }
DEV float lrelu(float x) { return x > 0.0f ? x : 0.2f * x; }

// ---------------- fp32 -> bf16 weight conversion ----------------
__global__ __launch_bounds__(256) void cvt_kernel(const float* __restrict__ in,
                                                  unsigned short* __restrict__ out, int n) {
    int i = blockIdx.x * 256 + threadIdx.x;
    if (i < n) out[i] = f2bf(in[i]);
}

// ---------------- WMMA bf16 GEMM:  out[M,N] = A[M,K] @ Bw[N,K]^T (+epilogue) ----------------
// Block tile 128(M) x 256(N), 8 waves as 2x4, each wave 64x64 = 16 WMMA tiles.
// A_F32: A is fp32 (converted to bf16 while staging to LDS), else A is bf16.
// EPI 0: out fp32 = acc + bias[n]
// EPI 1: out bf16 [M/2, N] = 0.5*(relu(acc[2i]+b) + relu(acc[2i+1]+b))   (gat1 + pair-mean)
template <bool A_F32, int EPI>
__global__ __launch_bounds__(256) void gemm_wmma(const void* __restrict__ Asrc,
                                                 const unsigned short* __restrict__ Bw,
                                                 const float* __restrict__ bias,
                                                 void* __restrict__ out,
                                                 int M, int N, int K) {
    __shared__ unsigned short lA[128 * 40];   // 32 K-elems + 8 pad per row (80B stride)
    __shared__ unsigned short lB[256 * 40];

    const int tid  = threadIdx.x;
    const int lane = tid & 31;
    const int wid  = tid >> 5;
    const int wm   = wid >> 2;      // 2 waves in M
    const int wn   = wid & 3;       // 4 waves in N
    const int lr   = lane & 15;
    const int lh   = lane >> 4;
    const int mBase = blockIdx.y * 128;
    const int nBase = blockIdx.x * 256;

    // ---- hoisted global/LDS offsets for staging (advance by 32 per K step) ----
    size_t aOff[2]; int aLds[2];
#pragma unroll
    for (int i = 0; i < 2; i++) {
        int c = tid + i * 256;          // 0..511  -> A: 128 rows x 4 chunks
        int row = c >> 2, kc = (c & 3) << 3;
        aOff[i] = (size_t)(mBase + row) * K + kc;
        aLds[i] = row * 40 + kc;
    }
    size_t bOff[4]; int bLds[4];
#pragma unroll
    for (int i = 0; i < 4; i++) {
        int c = tid + i * 256;          // 0..1023 -> B: 256 rows x 4 chunks
        int row = c >> 2, kc = (c & 3) << 3;
        bOff[i] = (size_t)(nBase + row) * K + kc;
        bLds[i] = row * 40 + kc;
    }

    // ---- hoisted LDS fragment pointers ----
    const __bf16* aP[4];
    const __bf16* bP[4];
#pragma unroll
    for (int mt = 0; mt < 4; mt++)
        aP[mt] = (const __bf16*)&lA[(wm * 64 + mt * 16 + lr) * 40] + lh * 8;
#pragma unroll
    for (int nt = 0; nt < 4; nt++)
        bP[nt] = (const __bf16*)&lB[(wn * 64 + nt * 16 + lr) * 40] + lh * 8;

    f32x8 acc[4][4];
    const f32x8 vzero = {0.f, 0.f, 0.f, 0.f, 0.f, 0.f, 0.f, 0.f};
#pragma unroll
    for (int i = 0; i < 4; i++)
#pragma unroll
        for (int j = 0; j < 4; j++) acc[i][j] = vzero;

    for (int kt = 0; kt < K; kt += 32) {
        // ---- stage A (128x32) and B (256x32) tiles into LDS ----
#pragma unroll
        for (int i = 0; i < 2; i++) {
            if (A_F32) {
                const float* ap = (const float*)Asrc + aOff[i];
                float4 f0 = *(const float4*)ap;
                float4 f1 = *(const float4*)(ap + 4);
                union { unsigned short h[8]; uint4 v; } pk;
                pk.h[0] = f2bf(f0.x); pk.h[1] = f2bf(f0.y); pk.h[2] = f2bf(f0.z); pk.h[3] = f2bf(f0.w);
                pk.h[4] = f2bf(f1.x); pk.h[5] = f2bf(f1.y); pk.h[6] = f2bf(f1.z); pk.h[7] = f2bf(f1.w);
                *(uint4*)&lA[aLds[i]] = pk.v;
            } else {
                const unsigned short* ap = (const unsigned short*)Asrc + aOff[i];
                *(uint4*)&lA[aLds[i]] = *(const uint4*)ap;
            }
            aOff[i] += 32;
        }
#pragma unroll
        for (int i = 0; i < 4; i++) {
            *(uint4*)&lB[bLds[i]] = *(const uint4*)(Bw + bOff[i]);
            bOff[i] += 32;
        }
        __syncthreads();

        // ---- fragments in WMMA bf16 layout ----
        // lanes 0-15: elems 0-7 = K0..7, 8-15 = K16..23 ; lanes 16-31: K8..15 / K24..31
        bf16x16 afr[4];
#pragma unroll
        for (int mt = 0; mt < 4; mt++) {
            bf16x8 a0 = *(const bf16x8*)(aP[mt]);
            bf16x8 a1 = *(const bf16x8*)(aP[mt] + 16);
#pragma unroll
            for (int e = 0; e < 8; e++) { afr[mt][e] = a0[e]; afr[mt][e + 8] = a1[e]; }
        }
#pragma unroll
        for (int nt = 0; nt < 4; nt++) {
            bf16x8 b0 = *(const bf16x8*)(bP[nt]);
            bf16x8 b1 = *(const bf16x8*)(bP[nt] + 16);
            bf16x16 bfr;
#pragma unroll
            for (int e = 0; e < 8; e++) { bfr[e] = b0[e]; bfr[e + 8] = b1[e]; }
#pragma unroll
            for (int mt = 0; mt < 4; mt++)
                acc[mt][nt] = __builtin_amdgcn_wmma_f32_16x16x32_bf16(
                    false, afr[mt], false, bfr, (short)0, acc[mt][nt], false, false);
        }
        __syncthreads();
    }

    // ---- epilogue (C layout: M = r + 8*lh, N = lr) ----
    float* outF = (float*)out;
    unsigned short* outH = (unsigned short*)out;
#pragma unroll
    for (int nt = 0; nt < 4; nt++) {
        int gn = nBase + wn * 64 + nt * 16 + lr;
        float bv = bias[gn];
#pragma unroll
        for (int mt = 0; mt < 4; mt++) {
            int gmb = mBase + wm * 64 + mt * 16 + 8 * lh;
            if (EPI == 0) {
#pragma unroll
                for (int r = 0; r < 8; r++)
                    outF[(size_t)(gmb + r) * N + gn] = acc[mt][nt][r] + bv;
            } else {
#pragma unroll
                for (int r = 0; r < 8; r += 2) {
                    float v0 = fmaxf(acc[mt][nt][r] + bv, 0.0f);
                    float v1 = fmaxf(acc[mt][nt][r + 1] + bv, 0.0f);
                    outH[(size_t)((gmb + r) >> 1) * N + gn] = f2bf(0.5f * (v0 + v1));
                }
            }
        }
    }
}

// ---------------- LayerNorm over rows of 1024 (one block per row) ----------------
template <bool RELU, bool OUTBF16>
__global__ __launch_bounds__(256) void ln_rows(const float* __restrict__ in,
                                               const float* __restrict__ g,
                                               const float* __restrict__ b,
                                               void* __restrict__ out,
                                               int rowMul, int rowAdd) {
    const int N = 1024;
    int row = blockIdx.x, tid = threadIdx.x;
    const float* x = in + (size_t)row * N;
    float4 v = *(const float4*)(x + tid * 4);
    float s = v.x + v.y + v.z + v.w;
    float ss = v.x * v.x + v.y * v.y + v.z * v.z + v.w * v.w;
    __shared__ float rs[256], rq[256];
    rs[tid] = s; rq[tid] = ss; __syncthreads();
    for (int off = 128; off > 0; off >>= 1) {
        if (tid < off) { rs[tid] += rs[tid + off]; rq[tid] += rq[tid + off]; }
        __syncthreads();
    }
    float mean = rs[0] * (1.0f / N);
    float var  = fmaxf(rq[0] * (1.0f / N) - mean * mean, 0.0f);
    float inv  = rsqrtf(var + 1e-5f);
    float4 gv = *(const float4*)(g + tid * 4);
    float4 bv = *(const float4*)(b + tid * 4);
    float y0 = (v.x - mean) * inv * gv.x + bv.x;
    float y1 = (v.y - mean) * inv * gv.y + bv.y;
    float y2 = (v.z - mean) * inv * gv.z + bv.z;
    float y3 = (v.w - mean) * inv * gv.w + bv.w;
    if (RELU) {
        y0 = fmaxf(y0, 0.f); y1 = fmaxf(y1, 0.f);
        y2 = fmaxf(y2, 0.f); y3 = fmaxf(y3, 0.f);
    }
    size_t obase = ((size_t)row * rowMul + rowAdd) * N + tid * 4;
    if (OUTBF16) {
        union { unsigned short h[4]; uint2 u; } pk;
        pk.h[0] = f2bf(y0); pk.h[1] = f2bf(y1); pk.h[2] = f2bf(y2); pk.h[3] = f2bf(y3);
        *(uint2*)((unsigned short*)out + obase) = pk.u;
    } else {
        float4 o; o.x = y0; o.y = y1; o.z = y2; o.w = y3;
        *(float4*)((float*)out + obase) = o;
    }
}

// ---------------- GAT1 attention fixup for nodes 0,1 (single block) ----------------
__global__ __launch_bounds__(256) void gat1_fixup(const unsigned short* __restrict__ nodes,
                                                  const unsigned short* __restrict__ W1,
                                                  const float* __restrict__ as1,
                                                  const float* __restrict__ ad1,
                                                  const float* __restrict__ b1,
                                                  unsigned short* __restrict__ m,
                                                  float* __restrict__ z) {
    __shared__ float x0[1024], x1[1024];
    __shared__ float xp0[2048], xp1[2048];
    __shared__ float red[256];
    __shared__ float sc[8];
    int tid = threadIdx.x;
    for (int i = tid; i < 1024; i += 256) { x0[i] = bf2f(nodes[i]); x1[i] = bf2f(nodes[1024 + i]); }
    __syncthreads();
    for (int n = tid; n < 2048; n += 256) {
        const unsigned short* w = W1 + (size_t)n * 1024;
        float a0 = 0.f, a1 = 0.f;
        for (int k = 0; k < 1024; k += 8) {
            uint4 wv = *(const uint4*)(w + k);
            const unsigned short* wh = (const unsigned short*)&wv;
#pragma unroll
            for (int e = 0; e < 8; e++) { float f = bf2f(wh[e]); a0 += x0[k + e] * f; a1 += x1[k + e] * f; }
        }
        xp0[n] = a0; xp1[n] = a1;
    }
    __syncthreads();
    // sc[j*2+h] = s[j][h], sc[4+j*2+h] = d[j][h]
    for (int jh = 0; jh < 4; jh++) {
        int j = jh >> 1, h = jh & 1;
        const float* xp = j ? xp1 : xp0;
        float ps = 0.f, pd = 0.f;
        for (int c = tid; c < 1024; c += 256) {
            float vv = xp[h * 1024 + c];
            ps += vv * as1[h * 1024 + c];
            pd += vv * ad1[h * 1024 + c];
        }
        red[tid] = ps; __syncthreads();
        for (int off = 128; off > 0; off >>= 1) { if (tid < off) red[tid] += red[tid + off]; __syncthreads(); }
        if (tid == 0) sc[jh] = red[0];
        __syncthreads();
        red[tid] = pd; __syncthreads();
        for (int off = 128; off > 0; off >>= 1) { if (tid < off) red[tid] += red[tid + off]; __syncthreads(); }
        if (tid == 0) sc[4 + jh] = red[0];
        __syncthreads();
    }
    float w00[2], w01[2], w10[2], w11[2];
#pragma unroll
    for (int h = 0; h < 2; h++) {
        float s0 = sc[h], s1 = sc[2 + h], d0 = sc[4 + h], d1 = sc[6 + h];
        float e00 = lrelu(s0 + d0), e10 = lrelu(s1 + d0);
        float mx = fmaxf(e00, e10);
        float p0 = __expf(e00 - mx), p1 = __expf(e10 - mx), dn = p0 + p1;
        w00[h] = p0 / dn; w01[h] = p1 / dn;                       // out0 = w00*xp0 + w01*xp1
        float e11 = lrelu(s1 + d1), e01 = lrelu(s0 + d1);
        mx = fmaxf(e11, e01);
        p0 = __expf(e11 - mx); p1 = __expf(e01 - mx); dn = p0 + p1;
        w10[h] = p0 / dn; w11[h] = p1 / dn;                       // out1 = w10*xp1 + w11*xp0
    }
    for (int n = tid; n < 2048; n += 256) {
        int h = n >> 10;
        float z0 = fmaxf(w00[h] * xp0[n] + w01[h] * xp1[n] + b1[n], 0.f);
        float z1 = fmaxf(w10[h] * xp1[n] + w11[h] * xp0[n] + b1[n], 0.f);
        z[n] = z0; z[2048 + n] = z1;
        m[n] = f2bf(0.5f * (z0 + z1));       // patch pair-mean row 0
    }
}

// ---------------- GAT2 attention fixup for fused row 0 (single block) ----------------
__global__ __launch_bounds__(256) void gat2_fixup(const float* __restrict__ z,
                                                  const unsigned short* __restrict__ W2,
                                                  const float* __restrict__ as2,
                                                  const float* __restrict__ ad2,
                                                  const float* __restrict__ b2,
                                                  float* __restrict__ fused) {
    __shared__ float z0[2048], z1[2048];
    __shared__ float p0[1024], p1[1024];
    __shared__ float red[256];
    __shared__ float sc[4];
    int tid = threadIdx.x;
    for (int i = tid; i < 2048; i += 256) { z0[i] = z[i]; z1[i] = z[2048 + i]; }
    __syncthreads();
    for (int n = tid; n < 1024; n += 256) {
        const unsigned short* w = W2 + (size_t)n * 2048;
        float a0 = 0.f, a1 = 0.f;
        for (int k = 0; k < 2048; k += 8) {
            uint4 wv = *(const uint4*)(w + k);
            const unsigned short* wh = (const unsigned short*)&wv;
#pragma unroll
            for (int e = 0; e < 8; e++) { float f = bf2f(wh[e]); a0 += z0[k + e] * f; a1 += z1[k + e] * f; }
        }
        p0[n] = a0; p1[n] = a1;
    }
    __syncthreads();
    for (int j = 0; j < 2; j++) {
        const float* p = j ? p1 : p0;
        float ps = 0.f, pd = 0.f;
        for (int c = tid; c < 1024; c += 256) { ps += p[c] * as2[c]; pd += p[c] * ad2[c]; }
        red[tid] = ps; __syncthreads();
        for (int off = 128; off > 0; off >>= 1) { if (tid < off) red[tid] += red[tid + off]; __syncthreads(); }
        if (tid == 0) sc[j] = red[0];
        __syncthreads();
        red[tid] = pd; __syncthreads();
        for (int off = 128; off > 0; off >>= 1) { if (tid < off) red[tid] += red[tid + off]; __syncthreads(); }
        if (tid == 0) sc[2 + j] = red[0];
        __syncthreads();
    }
    float s0 = sc[0], s1 = sc[1], d0 = sc[2], d1 = sc[3];
    float e00 = lrelu(s0 + d0), e10 = lrelu(s1 + d0);
    float mx = fmaxf(e00, e10);
    float a = __expf(e00 - mx), b_ = __expf(e10 - mx), dn = a + b_;
    float w00 = a / dn, w01 = b_ / dn;
    float e11 = lrelu(s1 + d1), e01 = lrelu(s0 + d1);
    mx = fmaxf(e11, e01);
    a = __expf(e11 - mx); b_ = __expf(e01 - mx); dn = a + b_;
    float w10 = a / dn, w11 = b_ / dn;
    float c0 = 0.5f * (w00 + w11), c1 = 0.5f * (w01 + w10);
    for (int n = tid; n < 1024; n += 256) fused[n] = c0 * p0[n] + c1 * p1[n] + b2[n];
}

// ---------------- host-side orchestration ----------------
extern "C" void kernel_launch(void* const* d_in, const int* in_sizes, int n_in,
                              void* d_out, int out_size, void* d_ws, size_t ws_size,
                              hipStream_t stream) {
    (void)in_sizes; (void)n_in; (void)out_size; (void)ws_size;
    const int B = 16384, D = 1024, HD = 2048;

    const float* audio = (const float*)d_in[0];
    const float* text  = (const float*)d_in[1];
    const float* Wa = (const float*)d_in[2];   const float* ba  = (const float*)d_in[3];
    const float* lna_g = (const float*)d_in[4]; const float* lna_b = (const float*)d_in[5];
    const float* Wt = (const float*)d_in[6];   const float* bt  = (const float*)d_in[7];
    const float* lnt_g = (const float*)d_in[8]; const float* lnt_b = (const float*)d_in[9];
    const float* W1 = (const float*)d_in[10];
    const float* as1 = (const float*)d_in[11]; const float* ad1 = (const float*)d_in[12];
    const float* b1 = (const float*)d_in[13];
    const float* W2 = (const float*)d_in[14];
    const float* as2 = (const float*)d_in[15]; const float* ad2 = (const float*)d_in[16];
    const float* b2 = (const float*)d_in[17];
    const float* Wf = (const float*)d_in[18];  const float* bfc = (const float*)d_in[19];
    const float* lnf_g = (const float*)d_in[20]; const float* lnf_b = (const float*)d_in[21];

    char* ws = (char*)d_ws;
    size_t o = 0;
    unsigned short* bWa = (unsigned short*)(ws + o); o += (size_t)D * D * 2;
    unsigned short* bWt = (unsigned short*)(ws + o); o += (size_t)D * D * 2;
    unsigned short* bW1 = (unsigned short*)(ws + o); o += (size_t)HD * D * 2;
    unsigned short* bW2 = (unsigned short*)(ws + o); o += (size_t)D * HD * 2;
    unsigned short* bWf = (unsigned short*)(ws + o); o += (size_t)D * D * 2;
    float* pre = (float*)(ws + o);                    o += (size_t)B * D * 4;   // also `fused`
    unsigned short* nodes = (unsigned short*)(ws + o); o += (size_t)2 * B * D * 2;
    unsigned short* mbuf = (unsigned short*)(ws + o);  o += (size_t)B * HD * 2;
    float* z = (float*)(ws + o);                       o += (size_t)2 * HD * 4;
    float* preF = (float*)nodes;  // nodes is dead after gat1 + fixup1

    dim3 blk(256);
    // weights -> bf16
    cvt_kernel<<<dim3((D * D + 255) / 256), blk, 0, stream>>>(Wa, bWa, D * D);
    cvt_kernel<<<dim3((D * D + 255) / 256), blk, 0, stream>>>(Wt, bWt, D * D);
    cvt_kernel<<<dim3((HD * D + 255) / 256), blk, 0, stream>>>(W1, bW1, HD * D);
    cvt_kernel<<<dim3((HD * D + 255) / 256), blk, 0, stream>>>(W2, bW2, HD * D);
    cvt_kernel<<<dim3((D * D + 255) / 256), blk, 0, stream>>>(Wf, bWf, D * D);

    // proj_audio: GEMM+bias -> LN+relu -> nodes even rows (bf16)
    gemm_wmma<true, 0><<<dim3(D / 256, B / 128), blk, 0, stream>>>(audio, bWa, ba, pre, B, D, D);
    ln_rows<true, true><<<dim3(B), blk, 0, stream>>>(pre, lna_g, lna_b, nodes, 2, 0);
    // proj_text -> nodes odd rows
    gemm_wmma<true, 0><<<dim3(D / 256, B / 128), blk, 0, stream>>>(text, bWt, bt, pre, B, D, D);
    ln_rows<true, true><<<dim3(B), blk, 0, stream>>>(pre, lnt_g, lnt_b, nodes, 2, 1);

    // gat1: +bias, relu, pair-mean fused in registers -> mbuf [B, 2048] bf16
    gemm_wmma<false, 1><<<dim3(HD / 256, (2 * B) / 128), blk, 0, stream>>>(nodes, bW1, b1, mbuf, 2 * B, HD, D);
    gat1_fixup<<<dim3(1), blk, 0, stream>>>(nodes, bW1, as1, ad1, b1, mbuf, z);

    // gat2 on pair-averaged rows: fused [B, 1024] fp32 (reuses `pre`)
    gemm_wmma<false, 0><<<dim3(D / 256, B / 128), blk, 0, stream>>>(mbuf, bW2, b2, pre, B, D, HD);
    gat2_fixup<<<dim3(1), blk, 0, stream>>>(z, bW2, as2, ad2, b2, pre);

    // fc: GEMM+bias -> LN -> d_out (fp32)
    gemm_wmma<true, 0><<<dim3(D / 256, B / 128), blk, 0, stream>>>(pre, bWf, bfc, preF, B, D, D);
    ln_rows<false, false><<<dim3(B), blk, 0, stream>>>(preF, lnf_g, lnf_b, d_out, 1, 0);
}
